// InterPConv_77163382440570
// MI455X (gfx1250) — matrix operand order repaired and checked by reading, not compiled, not run on hardware
//
#include <hip/hip_runtime.h>
#include <hip/hip_bf16.h>

// ---------------------------------------------------------------------------
// InterpConv on MI455X (gfx1250, wave32)
//   BS=8, N=2048, CIN=128, COUT=256, KS=LS=16
// Pipeline:
//   K0: conv_w f32 -> bf16 (already [COUT, CIN*KS] = B^T for the GEMM)
//   K1: per-batch KNN (xyz staged in LDS, register insertion sort, top-15)
//   K2: BN partial sums of h = rel@w1+b1 (wave shfl reduce -> block partials)
//   K3: BN finalize (mean, rstd) -- fixed order, deterministic
//   K4: alpha (MLP+BN+softmax) + neighbor gather + agg BMM, store agg as bf16
//   K5: GEMM out[16384,256] = agg(bf16) x conv_w^T(bf16) + bias
//       v_wmma_f32_16x16x32_bf16, 128x128 block tile, 8 waves (2Mx4N),
//       double-buffered LDS filled by GLOBAL_LOAD_ASYNC_TO_LDS_B128
//       (ASYNCcnt) so the async engine prefetches tile k+1 during WMMA on k.
// ---------------------------------------------------------------------------

typedef __attribute__((ext_vector_type(16))) __bf16 v16bf;
typedef __attribute__((ext_vector_type(8)))  float  v8f;

#define BS_   8
#define N_    2048
#define CIN_  128
#define COUT_ 256
#define KS_   16
#define LS_   16
#define NPTS  (BS_ * N_)          // 16384
#define NROWS (NPTS * LS_)        // 262144
#define KDIM  (CIN_ * KS_)        // 2048

// ---------------- workspace layout (bytes), total ~69.5 MB ------------------
static constexpr size_t WS_KNN   = 0;         // int [NPTS*16]
static constexpr size_t WS_PSUM  = 1048576;   // f32 [1024*32]
static constexpr size_t WS_PSQ   = 1179648;   // f32 [1024*32]
static constexpr size_t WS_STAT  = 1310720;   // f32 [64]
static constexpr size_t WS_AGG   = 1310976;   // bf16[NPTS*KDIM]
static constexpr size_t WS_CONVW = 68419840;  // bf16[COUT_*KDIM]

// ---------------------------------------------------------------------------
__global__ void k0_cvt_w(const float* __restrict__ w, __bf16* __restrict__ o) {
    int i = blockIdx.x * 256 + threadIdx.x;
    o[i] = (__bf16)w[i];
}

// ---------------------------------------------------------------------------
// K1: KNN. grid = 64 blocks (b = blk/8, chunk = blk%8), 256 threads.
__global__ __launch_bounds__(256) void k1_knn(const float* __restrict__ xyz,
                                              int* __restrict__ knn) {
    __shared__ float sxyz[N_ * 3];               // 24 KB of 320 KB WGP LDS
    const int blk = blockIdx.x;
    const int b = blk >> 3, chunk = blk & 7;
    const int tid = threadIdx.x;
    for (int i = tid; i < N_ * 3; i += 256)
        sxyz[i] = xyz[b * N_ * 3 + i];
    __syncthreads();

    const int q = chunk * 256 + tid;
    const float qx = sxyz[q * 3 + 0];
    const float qy = sxyz[q * 3 + 1];
    const float qz = sxyz[q * 3 + 2];

    float bd[15]; int bi[15];
#pragma unroll
    for (int s = 0; s < 15; ++s) { bd[s] = 3.4e38f; bi[s] = 0; }

    for (int j = 0; j < N_; ++j) {
        const float dx = sxyz[j * 3 + 0] - qx;
        const float dy = sxyz[j * 3 + 1] - qy;
        const float dz = sxyz[j * 3 + 2] - qz;
        float dd = fmaf(dx, dx, fmaf(dy, dy, dz * dz));
        if (dd < 1e-8f) continue;                // mask self / duplicates
        if (dd < bd[14]) {
            float d = dd; int id = j;
#pragma unroll
            for (int s = 0; s < 15; ++s) {       // unrolled sorted insert
                if (d < bd[s]) {
                    float td = bd[s]; int ti = bi[s];
                    bd[s] = d; bi[s] = id; d = td; id = ti;
                }
            }
        }
    }
    const int p = b * N_ + q;
    knn[p * LS_ + 0] = q;                        // self first
#pragma unroll
    for (int s = 0; s < 15; ++s) knn[p * LS_ + 1 + s] = bi[s];
}

// ---------------------------------------------------------------------------
// K2: per-block partial sums of h and h^2. grid = 1024 x 256 (one row/thread)
__global__ __launch_bounds__(256) void k2_bnpart(
    const float* __restrict__ xyz, const int* __restrict__ knn,
    const float* __restrict__ w1, const float* __restrict__ b1,
    float* __restrict__ psum, float* __restrict__ psq) {
    __shared__ float wsum[8 * 32];
    __shared__ float wsq[8 * 32];
    const int tid = threadIdx.x;
    const int r = blockIdx.x * 256 + tid;
    const int p = r >> 4;
    const int b = p >> 11, n = p & (N_ - 1);
    const int j = knn[r];

    const float rx = xyz[(b * N_ + j) * 3 + 0] - xyz[(b * N_ + n) * 3 + 0];
    const float ry = xyz[(b * N_ + j) * 3 + 1] - xyz[(b * N_ + n) * 3 + 1];
    const float rz = xyz[(b * N_ + j) * 3 + 2] - xyz[(b * N_ + n) * 3 + 2];

    const int lane = tid & 31, wid = tid >> 5;
#pragma unroll
    for (int c = 0; c < 32; ++c) {
        float h = fmaf(rx, w1[c], fmaf(ry, w1[32 + c], fmaf(rz, w1[64 + c], b1[c])));
        float s = h, qq = h * h;
#pragma unroll
        for (int off = 16; off > 0; off >>= 1) { // wave32 tree reduce
            s  += __shfl_xor(s,  off, 32);
            qq += __shfl_xor(qq, off, 32);
        }
        if (lane == 0) { wsum[wid * 32 + c] = s; wsq[wid * 32 + c] = qq; }
    }
    __syncthreads();
    if (tid < 32) {                              // fixed order: deterministic
        float s = 0.f, qq = 0.f;
#pragma unroll
        for (int w = 0; w < 8; ++w) { s += wsum[w * 32 + tid]; qq += wsq[w * 32 + tid]; }
        psum[blockIdx.x * 32 + tid] = s;
        psq [blockIdx.x * 32 + tid] = qq;
    }
}

// K3: finalize mean / rstd. 1 block, 32 threads, fixed order.
__global__ void k3_bnfin(const float* __restrict__ psum,
                         const float* __restrict__ psq,
                         float* __restrict__ stat) {
    const int c = threadIdx.x;
    float s = 0.f, qq = 0.f;
    for (int i = 0; i < 1024; ++i) { s += psum[i * 32 + c]; qq += psq[i * 32 + c]; }
    const float inv = 1.0f / (float)NROWS;
    const float mu = s * inv;
    const float var = fmaxf(qq * inv - mu * mu, 0.0f);
    stat[c]      = mu;
    stat[32 + c] = __frsqrt_rn(var + 1e-5f);
}

// ---------------------------------------------------------------------------
// K4: alpha MLP + gather + agg BMM. grid = 16384 blocks x 128 threads.
__global__ __launch_bounds__(128) void k4_agg(
    const float* __restrict__ xyz, const float* __restrict__ data,
    const int* __restrict__ knn,
    const float* __restrict__ w1, const float* __restrict__ b1,
    const float* __restrict__ gamma, const float* __restrict__ beta,
    const float* __restrict__ w2, const float* __restrict__ b2,
    const float* __restrict__ stat, __bf16* __restrict__ agg) {
    const int p = blockIdx.x;
    const int b = p >> 11, n = p & (N_ - 1);
    const int tid = threadIdx.x;

    __shared__ int   sidx[LS_];
    __shared__ float salpha[LS_][KS_];
    if (tid < LS_) sidx[tid] = knn[p * LS_ + tid];
    __syncthreads();

    if (tid < LS_) {
        const int j = sidx[tid];
        const float rx = xyz[(b * N_ + j) * 3 + 0] - xyz[(b * N_ + n) * 3 + 0];
        const float ry = xyz[(b * N_ + j) * 3 + 1] - xyz[(b * N_ + n) * 3 + 1];
        const float rz = xyz[(b * N_ + j) * 3 + 2] - xyz[(b * N_ + n) * 3 + 2];
        float h[32];
#pragma unroll
        for (int c = 0; c < 32; ++c) {
            float v = fmaf(rx, w1[c], fmaf(ry, w1[32 + c], fmaf(rz, w1[64 + c], b1[c])));
            v = (v - stat[c]) * stat[32 + c] * gamma[c] + beta[c];
            h[c] = fmaxf(v, 0.0f);               // BN + ReLU
        }
        float t[KS_];
#pragma unroll
        for (int k = 0; k < KS_; ++k) {
            float s = b2[k];
#pragma unroll
            for (int c = 0; c < 32; ++c) s = fmaf(h[c], w2[c * KS_ + k], s);
            t[k] = s;
        }
        float mx = t[0];
#pragma unroll
        for (int k = 1; k < KS_; ++k) mx = fmaxf(mx, t[k]);
        float sum = 0.f;
#pragma unroll
        for (int k = 0; k < KS_; ++k) { t[k] = __expf(t[k] - mx); sum += t[k]; }
        const float inv = 1.0f / sum;
#pragma unroll
        for (int k = 0; k < KS_; ++k) salpha[tid][k] = t[k] * inv;
    }
    __syncthreads();

    float rd[LS_];
#pragma unroll
    for (int l = 0; l < LS_; ++l)
        rd[l] = data[(size_t)(b * N_ + sidx[l]) * CIN_ + tid];

    __bf16 tmp[KS_];
#pragma unroll
    for (int k = 0; k < KS_; ++k) {
        float s = 0.f;
#pragma unroll
        for (int l = 0; l < LS_; ++l) s = fmaf(rd[l], salpha[l][k], s);
        tmp[k] = (__bf16)s;
    }
    __builtin_memcpy(agg + (size_t)p * KDIM + tid * KS_, tmp, 32);  // 32B store
}

// ---------------------------------------------------------------------------
// K5: GEMM  C[16384,256] = A[16384,2048]bf16 x Bt[256,2048]bf16^T + bias
//   Block tile 128(M) x 128(N), 8 waves (2Mx4N), wave tile 64x32 = 4x2 WMMA.
//   Double-buffered LDS; tiles fetched with GLOBAL_LOAD_ASYNC_TO_LDS_B128.
#define LDA 40                     // shorts; 80B row pitch, 16B aligned
#define TILE_E (128 * LDA)         // elements per (A or B) tile region
#define BUF_E  (2 * TILE_E)        // elements per buffer (A tile + B tile)
__global__ __launch_bounds__(256) void k5_gemm(
    const __bf16* __restrict__ A, const __bf16* __restrict__ Bt,
    const float* __restrict__ bias, float* __restrict__ out) {
    // [2 buffers][A tile 128x40 | B tile 128x40] bf16 = 40960 B
    __shared__ __align__(16) __bf16 smem[2 * BUF_E];

    const int tid = threadIdx.x;
    const int m0 = blockIdx.x * 128;
    const int n0 = blockIdx.y * 128;
    const int w = tid >> 5, lane = tid & 31;
    const int wm = w & 1, wn = w >> 1;           // 2 x 4 wave grid
    const int lrow = lane & 15, lhalf = lane >> 4;

    const v8f vzero = {0.f, 0.f, 0.f, 0.f, 0.f, 0.f, 0.f, 0.f};
    v8f acc[4][2];
#pragma unroll
    for (int mi = 0; mi < 4; ++mi)
#pragma unroll
        for (int ni = 0; ni < 2; ++ni) acc[mi][ni] = vzero;

    // --- async tile staging: each thread owns one 32B chunk of each tile ----
    const int r = tid >> 1;                      // row 0..127 (M for A, N for B)
    const int h = (tid & 1) * 16;                // k-element offset 0 or 16
    // LDS byte offsets: low 32 bits of the generic (flat) pointer are the LDS
    // address (LDS aperture: addr[31:0] = LDS offset).
    const unsigned ldsA0 = (unsigned)(uintptr_t)&smem[r * LDA + h];
    const unsigned ldsB0 = (unsigned)(uintptr_t)&smem[TILE_E + r * LDA + h];
    const unsigned aOffB = (unsigned)(((size_t)(m0 + r) * KDIM + h) * 2);
    const unsigned bOffB = (unsigned)(((size_t)(n0 + r) * KDIM + h) * 2);
    const unsigned long long aBase = (unsigned long long)(uintptr_t)A;
    const unsigned long long bBase = (unsigned long long)(uintptr_t)Bt;

#define ISSUE_TILE(bufsel, kk)                                                 \
    do {                                                                       \
        unsigned la = ldsA0 + (bufsel) * (BUF_E * 2);                          \
        unsigned lb = ldsB0 + (bufsel) * (BUF_E * 2);                          \
        unsigned va = aOffB + (kk) * 2;                                        \
        unsigned vb = bOffB + (kk) * 2;                                        \
        asm volatile(                                                          \
            "global_load_async_to_lds_b128 %0, %2, %4 offset:0\n\t"            \
            "global_load_async_to_lds_b128 %0, %2, %4 offset:16\n\t"           \
            "global_load_async_to_lds_b128 %1, %3, %5 offset:0\n\t"            \
            "global_load_async_to_lds_b128 %1, %3, %5 offset:16"               \
            :                                                                  \
            : "v"(la), "v"(lb), "v"(va), "v"(vb), "s"(aBase), "s"(bBase)       \
            : "memory");                                                       \
    } while (0)

    ISSUE_TILE(0, 0);                            // prologue: fill buffer 0

    for (int kk = 0; kk < KDIM; kk += 32) {
        const int cur = (kk >> 5) & 1;
        // my async loads for buffer `cur` must have landed; then barrier so
        // everyone's have (each wave waits its own before signaling).
        asm volatile("s_wait_asynccnt 0x0" ::: "memory");
        __syncthreads();
        if (kk + 32 < KDIM) ISSUE_TILE(cur ^ 1, kk + 32);   // overlap w/ WMMA

        const __bf16* Ab = &smem[cur * BUF_E];
        const __bf16* Bb = &smem[cur * BUF_E + TILE_E];

        // A fragment (16x32 bf16): lane = row (L&15); lanes0-15 K=0..7,16..23;
        //                          lanes16-31 K=8..15,24..31
        v16bf af[4], bfr[2];
#pragma unroll
        for (int mi = 0; mi < 4; ++mi) {
            const __bf16* ap = &Ab[(wm * 64 + mi * 16 + lrow) * LDA];
#pragma unroll
            for (int e = 0; e < 8; ++e) af[mi][e]     = ap[lhalf * 8 + e];
#pragma unroll
            for (int e = 0; e < 8; ++e) af[mi][8 + e] = ap[16 + lhalf * 8 + e];
        }
        // B fragment (32x16 bf16): lane = col (L&15); lanes0-15 K=0..7,16..23;
        //                          lanes16-31 K=8..15,24..31 (B tile is [n][k])
#pragma unroll
        for (int ni = 0; ni < 2; ++ni) {
            const __bf16* bp = &Bb[(wn * 32 + ni * 16 + lrow) * LDA];
#pragma unroll
            for (int e = 0; e < 8; ++e) bfr[ni][e]     = bp[lhalf * 8 + e];
#pragma unroll
            for (int e = 0; e < 8; ++e) bfr[ni][8 + e] = bp[16 + lhalf * 8 + e];
        }
#pragma unroll
        for (int mi = 0; mi < 4; ++mi)
#pragma unroll
            for (int ni = 0; ni < 2; ++ni)
                acc[mi][ni] = __builtin_amdgcn_wmma_f32_16x16x32_bf16(
                    false, af[mi], false, bfr[ni], (short)0, acc[mi][ni],
                    false, false);
    }
#undef ISSUE_TILE

    // epilogue: C/D layout -> VGPR r: M = r + 8*(L>>4), N = L&15
#pragma unroll
    for (int mi = 0; mi < 4; ++mi) {
#pragma unroll
        for (int ni = 0; ni < 2; ++ni) {
            const int colg = n0 + wn * 32 + ni * 16 + lrow;
            const float bv = bias[colg];
#pragma unroll
            for (int rr = 0; rr < 8; ++rr) {
                const int rowg = m0 + wm * 64 + mi * 16 + lhalf * 8 + rr;
                out[(size_t)rowg * COUT_ + colg] = acc[mi][ni][rr] + bv;
            }
        }
    }
}
#undef LDA
#undef TILE_E
#undef BUF_E

// ---------------------------------------------------------------------------
extern "C" void kernel_launch(void* const* d_in, const int* in_sizes, int n_in,
                              void* d_out, int out_size, void* d_ws, size_t ws_size,
                              hipStream_t stream) {
    const float* xyz    = (const float*)d_in[0];
    const float* data   = (const float*)d_in[1];
    const float* w1     = (const float*)d_in[2];
    const float* b1     = (const float*)d_in[3];
    const float* gamma  = (const float*)d_in[4];
    const float* beta   = (const float*)d_in[5];
    const float* w2     = (const float*)d_in[6];
    const float* b2     = (const float*)d_in[7];
    const float* conv_w = (const float*)d_in[8];
    const float* conv_b = (const float*)d_in[9];

    char* ws = (char*)d_ws;                       // needs ~70 MB
    int*    knn   = (int*)   (ws + WS_KNN);
    float*  psum  = (float*) (ws + WS_PSUM);
    float*  psq   = (float*) (ws + WS_PSQ);
    float*  stat  = (float*) (ws + WS_STAT);
    __bf16* agg   = (__bf16*)(ws + WS_AGG);
    __bf16* convw = (__bf16*)(ws + WS_CONVW);

    float* out_xyz = (float*)d_out;               // first 49152 floats
    float* out_fea = (float*)d_out + BS_ * N_ * 3;

    hipMemcpyAsync(out_xyz, xyz, (size_t)BS_ * N_ * 3 * sizeof(float),
                   hipMemcpyDeviceToDevice, stream);

    k0_cvt_w<<<dim3(COUT_ * KDIM / 256), dim3(256), 0, stream>>>(conv_w, convw);
    k1_knn  <<<dim3(BS_ * 8),            dim3(256), 0, stream>>>(xyz, knn);
    k2_bnpart<<<dim3(NROWS / 256),       dim3(256), 0, stream>>>(xyz, knn, w1, b1, psum, psq);
    k3_bnfin <<<dim3(1),                 dim3(32),  0, stream>>>(psum, psq, stat);
    k4_agg   <<<dim3(NPTS),              dim3(128), 0, stream>>>(
        xyz, data, knn, w1, b1, gamma, beta, w2, b2, stat, agg);
    k5_gemm  <<<dim3(NPTS / 128, COUT_ / 128), dim3(256), 0, stream>>>(
        agg, convw, conv_b, out_fea);
}
